// SequenceTagger_85547158602188
// MI455X (gfx1250) — compile-verified, matrix operand order checked by reading, not run on hardware
//
#include <hip/hip_runtime.h>
#include <hip/hip_bf16.h>
#include <stdint.h>

// ---------------------------------------------------------------------------
// Problem constants (from reference)
// ---------------------------------------------------------------------------
#define SL 256
#define BS 64
#define DIM 1024
#define HID 512
#define GG  2048           // 4*H
#define NT  17             // tags
#define SLBS (SL*BS)       // 16384
#define BN_EPS 1e-5f

// ---------------------------------------------------------------------------
// WMMA fragment types (gfx1250, wave32)
// ---------------------------------------------------------------------------
typedef __attribute__((ext_vector_type(16))) __bf16 v16bf;
typedef __attribute__((ext_vector_type(8)))  float  v8f;
typedef unsigned int v4u __attribute__((ext_vector_type(4)));
typedef int          v4i __attribute__((ext_vector_type(4)));
typedef int          v8i __attribute__((ext_vector_type(8)));

union Frag16 { uint4 q[2]; unsigned int u[8]; v16bf v; };

__device__ __forceinline__ unsigned short f2bf(float f) {
    unsigned int u = __float_as_uint(f);
    return (unsigned short)((u + 0x7FFFu + ((u >> 16) & 1u)) >> 16);
}
__device__ __forceinline__ float sigf(float x) { return 1.0f / (1.0f + __expf(-x)); }

#if __has_builtin(__builtin_amdgcn_tensor_load_to_lds) && __has_builtin(__builtin_amdgcn_s_wait_tensorcnt)
#define HAVE_TDM 1
#else
#define HAVE_TDM 0
#endif

#if HAVE_TDM
// Issue one TDM 2D tile load (Global -> LDS). EXEC is ignored by the tensor op,
// so executing this once in wave 0 issues exactly one DMA.
// D# bit layout per CDNA5 ISA ch.8: group0 {count, lds_addr, global_addr, type=2},
// group1 {data_size=1 (2-byte elems), tensor_dim0/1, tile_dim0/1, dim0 stride}.
// This toolchain exposes the 6-arg builtin:
//   (v4u g0, v8i g1, v4i g2, v4i g3, v8i g4, i32 cpol)
__device__ __forceinline__ void tdm_load_2d(unsigned lds_addr, const void* gptr,
                                            unsigned tensor_w, unsigned tensor_h,
                                            unsigned stride_elems,
                                            unsigned tile_w, unsigned tile_h)
{
    unsigned long long ga = (unsigned long long)(uintptr_t)gptr;
    v4u g0;
    g0.x = 1u;                                              // count=1 (valid user D#)
    g0.y = lds_addr;                                        // LDS byte address
    g0.z = (unsigned)(ga & 0xFFFFFFFFu);                    // global addr [95:64]
    g0.w = (unsigned)((ga >> 32) & 0x01FFFFFFu)             // global addr [120:96]
         | 0x80000000u;                                     // type=2 ("image") [127:126]
    v8i g1;
    g1[0] = (int)(1u << 16);                                // wg_mask=0, data_size=1 (2B)
    g1[1] = (int)(tensor_w << 16);                          // tensor_dim0[15:0]  @ [63:48]
    g1[2] = (int)((tensor_w >> 16) | (tensor_h << 16));     // dim0[31:16] | dim1[15:0]
    g1[3] = (int)((tensor_h >> 16) | (tile_w << 16));       // dim1[31:16] | tile_dim0
    g1[4] = (int)tile_h;                                    // tile_dim1 (tile_dim2=0)
    g1[5] = (int)stride_elems;                              // tensor_dim0_stride[31:0]
    g1[6] = 0;                                              // stride hi | dim1_stride lo
    g1[7] = 0;
    v4i gz4 = {0, 0, 0, 0};                                 // groups 2/3 unused (2D tile)
    v8i gz8 = {0, 0, 0, 0, 0, 0, 0, 0};                     // extra group (unused)
    __builtin_amdgcn_tensor_load_to_lds(g0, g1, gz4, gz4, gz8, 0);
}
#endif

// ---------------------------------------------------------------------------
// 0) f32 -> bf16 conversion (word_input, weight matrices)
// ---------------------------------------------------------------------------
__global__ void f32_to_bf16_k(const float* __restrict__ src, unsigned short* __restrict__ dst, int n) {
    for (int i = blockIdx.x * blockDim.x + threadIdx.x; i < n; i += gridDim.x * blockDim.x)
        dst[i] = f2bf(src[i]);
}

// ---------------------------------------------------------------------------
// 1) xg GEMM:  XG[m, g] = sum_d X[m,d] * W[g,d] + bih[g] + bhh[g]
//    M=16384, N=2048, K=1024.  128x128 block tile, 8 waves.
//    A/B slabs staged to LDS by the Tensor Data Mover (8 KB tiles), WMMA core.
// ---------------------------------------------------------------------------
__global__ void __launch_bounds__(256)
xg_gemm_k(const unsigned short* __restrict__ Xbf,   // 16384 x 1024
          const unsigned short* __restrict__ Wbf,   // 2048 x 1024
          const float* __restrict__ bih, const float* __restrict__ bhh,
          float* __restrict__ XG)                   // 16384 x 2048
{
    __shared__ unsigned int As[128 * 16];   // 128 rows x 32 K (bf16) = 8 KB, row pitch 64 B
    __shared__ unsigned int Bs[128 * 16];   // 128 cols x 32 K        = 8 KB

    const int m0 = blockIdx.y * 128;
    const int n0 = blockIdx.x * 128;
    const int tid  = threadIdx.x;
    const int wid  = tid >> 5;              // wave id 0..7 -> M tile
    const int lane = tid & 31;
    const int lr = lane & 15, lh = lane >> 4;

    v8f zero = {0.f,0.f,0.f,0.f,0.f,0.f,0.f,0.f};
    v8f acc[8];
    for (int nt = 0; nt < 8; ++nt) acc[nt] = zero;

    for (int k0 = 0; k0 < DIM; k0 += 32) {
#if HAVE_TDM
        if (wid == 0) {   // one wave issues both DMAs; tensor op ignores EXEC
            tdm_load_2d((unsigned)(uintptr_t)(void*)As,
                        Xbf + (size_t)m0 * DIM + k0, DIM, SLBS, DIM, 32, 128);
            tdm_load_2d((unsigned)(uintptr_t)(void*)Bs,
                        Wbf + (size_t)n0 * DIM + k0, DIM, GG, DIM, 32, 128);
            __builtin_amdgcn_s_wait_tensorcnt(0);
        }
#else
        for (int i = 0; i < 8; ++i) {       // fallback: cooperative dword staging
            int e = tid + i * 256;
            int r = e >> 4, c = e & 15;
            As[e] = ((const unsigned int*)(Xbf + (size_t)(m0 + r) * DIM + k0))[c];
            Bs[e] = ((const unsigned int*)(Wbf + (size_t)(n0 + r) * DIM + k0))[c];
        }
        if (k0 + 32 < DIM) {
            __builtin_prefetch(Xbf + (size_t)(m0 + (tid >> 1)) * DIM + k0 + 32, 0, 1);
            __builtin_prefetch(Wbf + (size_t)(n0 + (tid >> 1)) * DIM + k0 + 32, 0, 1);
        }
#endif
        __syncthreads();

        Frag16 a;
        {
            const uint4* ap = (const uint4*)&As[(wid * 16 + lr) * 16];
            a.q[0] = ap[lh];        // bytes lh*16      : K = lh*8 + {0,2,4,6} pairs
            a.q[1] = ap[2 + lh];    // bytes 32 + lh*16 : K = 16 + lh*8 + ...
        }
        for (int nt = 0; nt < 8; ++nt) {
            Frag16 b;
            const uint4* bp = (const uint4*)&Bs[(nt * 16 + lr) * 16];
            b.q[0] = bp[lh];
            b.q[1] = bp[2 + lh];
            acc[nt] = __builtin_amdgcn_wmma_f32_16x16x32_bf16(
                false, a.v, false, b.v, (short)0, acc[nt], false, false);
        }
        __syncthreads();
    }

    for (int nt = 0; nt < 8; ++nt) {
        int n = n0 + nt * 16 + lr;
        float bias = bih[n] + bhh[n];
        for (int v = 0; v < 8; ++v) {
            int m = m0 + wid * 16 + v + lh * 8;
            XG[(size_t)m * GG + n] = acc[nt][v] + bias;
        }
    }
}

// ---------------------------------------------------------------------------
// 2) LSTM scan, one direction per block.  1024 threads = 32 waves.
//    Per step: G = XG[s] + h_prev @ w_hh^T via WMMA (M=64, N=2048, K=512),
//    fused sigmoid/tanh cell update.  h_prev in LDS (bf16), c in registers,
//    w_hh L2-resident; fragments loaded as 2x b128.
// ---------------------------------------------------------------------------
__global__ void __launch_bounds__(1024)
lstm_scan_k(const float* __restrict__ xgf, const float* __restrict__ xgb,
            const unsigned short* __restrict__ whhf, const unsigned short* __restrict__ whhb,
            float* __restrict__ hf, float* __restrict__ hb)
{
    __shared__ unsigned short hbuf[64 * 512];   // 64 KB: h_prev in bf16, row pitch 1 KB

    const int dir = blockIdx.x;
    const float* __restrict__ xg = dir ? xgb : xgf;
    const unsigned short* __restrict__ whh = dir ? whhb : whhf;
    float* __restrict__ hs = dir ? hb : hf;

    const int tid  = threadIdx.x;
    const int wid  = tid >> 5, lane = tid & 31;
    const int lr = lane & 15, lh = lane >> 4;
    const int mt = wid & 3;        // batch tile (rows mt*16..+15)
    const int c8 = wid >> 2;       // 0..7, h-column group (4 tiles each)

    for (int i = tid; i < 64 * 512; i += 1024) hbuf[i] = 0;

    float cst[4][8];
    for (int q = 0; q < 4; ++q)
        for (int v = 0; v < 8; ++v) cst[q][v] = 0.f;

    v8f zero = {0.f,0.f,0.f,0.f,0.f,0.f,0.f,0.f};
    __syncthreads();

    for (int t = 0; t < SL; ++t) {
        int s = dir ? (SL - 1 - t) : t;

        v8f acc[4][4];
        for (int q = 0; q < 4; ++q)
            for (int g = 0; g < 4; ++g) acc[q][g] = zero;

        for (int k0 = 0; k0 < HID; k0 += 32) {
            Frag16 a;                               // h_prev tile, shared by this wave's 16 tiles
            const uint4* hp = (const uint4*)&hbuf[(mt * 16 + lr) * 512 + k0];
            a.q[0] = hp[lh];
            a.q[1] = hp[2 + lh];
            for (int q = 0; q < 4; ++q) {
                int ht = c8 * 4 + q;                // h tile 0..31
                for (int g = 0; g < 4; ++g) {
                    int gcol = (g * 32 + ht) * 16 + lr;   // gate column for this lane
                    Frag16 b;
                    const uint4* bp = (const uint4*)(whh + (size_t)gcol * HID + k0);
                    b.q[0] = bp[lh];
                    b.q[1] = bp[2 + lh];
                    acc[q][g] = __builtin_amdgcn_wmma_f32_16x16x32_bf16(
                        false, a.v, false, b.v, (short)0, acc[q][g], false, false);
                }
            }
        }
        __syncthreads();                            // all reads of hbuf done

        for (int q = 0; q < 4; ++q) {
            int hcol = (c8 * 4 + q) * 16 + lr;
            for (int v = 0; v < 8; ++v) {
                int bidx = mt * 16 + v + lh * 8;
                size_t row = (size_t)s * BS + bidx;
                const float* xrow = xg + row * GG;
                float gi = acc[q][0][v] + xrow[0 * HID + hcol];
                float gf = acc[q][1][v] + xrow[1 * HID + hcol];
                float gc = acc[q][2][v] + xrow[2 * HID + hcol];
                float go = acc[q][3][v] + xrow[3 * HID + hcol];
                float c  = sigf(gf) * cst[q][v] + sigf(gi) * tanhf(gc);
                cst[q][v] = c;
                float h  = sigf(go) * tanhf(c);
                hs[row * HID + hcol] = h;
                hbuf[bidx * 512 + hcol] = f2bf(h);
            }
        }
        __syncthreads();                            // h_prev ready for next step
    }
}

// ---------------------------------------------------------------------------
// 3) BatchNorm statistics over rows of concat(hf, hb): one block per column.
// ---------------------------------------------------------------------------
__global__ void bn_stats_k(const float* __restrict__ hf, const float* __restrict__ hb,
                           float* __restrict__ mean, float* __restrict__ var)
{
    __shared__ float s1[256], s2[256];
    int col = blockIdx.x;                 // 0..1023
    const float* base = (col < HID) ? (hf + col) : (hb + col - HID);
    float a = 0.f, b = 0.f;
    for (int m = threadIdx.x; m < SLBS; m += 256) {
        float v = base[(size_t)m * HID];
        a += v; b += v * v;
    }
    s1[threadIdx.x] = a; s2[threadIdx.x] = b;
    __syncthreads();
    for (int off = 128; off; off >>= 1) {
        if (threadIdx.x < off) { s1[threadIdx.x] += s1[threadIdx.x + off]; s2[threadIdx.x] += s2[threadIdx.x + off]; }
        __syncthreads();
    }
    if (threadIdx.x == 0) {
        float mu = s1[0] / (float)SLBS;
        mean[col] = mu;
        var[col]  = s2[0] / (float)SLBS - mu * mu;
    }
}

// ---------------------------------------------------------------------------
// 4) Fold BN into linear: Wp[t][j] = W[t][j]*g[j]*rstd[j];
//    bp[t] = b[t] + sum_j W[t][j]*(beta[j] - mean[j]*g[j]*rstd[j])
// ---------------------------------------------------------------------------
__global__ void fold_bn_k(const float* __restrict__ lw, const float* __restrict__ lb,
                          const float* __restrict__ gamma, const float* __restrict__ beta,
                          const float* __restrict__ mean, const float* __restrict__ var,
                          float* __restrict__ Wp, float* __restrict__ bp)
{
    __shared__ float red[256];
    int t = blockIdx.x;
    float part = 0.f;
    for (int j = threadIdx.x; j < 2 * HID; j += 256) {
        float w  = lw[t * 2 * HID + j];
        float sc = gamma[j] * rsqrtf(var[j] + BN_EPS);
        Wp[t * 2 * HID + j] = w * sc;
        part += w * (beta[j] - mean[j] * sc);
    }
    red[threadIdx.x] = part;
    __syncthreads();
    for (int off = 128; off; off >>= 1) {
        if (threadIdx.x < off) red[threadIdx.x] += red[threadIdx.x + off];
        __syncthreads();
    }
    if (threadIdx.x == 0) bp[t] = lb[t] + red[0];
}

// ---------------------------------------------------------------------------
// 5) Emission logits: E[b][s][t] = lstm_out[m] . Wp[t] + bp[t], m = s*BS+b
// ---------------------------------------------------------------------------
__global__ void __launch_bounds__(256)
logits_k(const float* __restrict__ hf, const float* __restrict__ hb,
         const float* __restrict__ Wp, const float* __restrict__ bp,
         float* __restrict__ E)
{
    __shared__ float xr[8][2 * HID];
    int m0 = blockIdx.x * 8;
    int tid = threadIdx.y * 32 + threadIdx.x;
    for (int i = tid; i < 8 * 2 * HID; i += 256) {
        int r = i >> 10, j = i & 1023;
        size_t m = (size_t)(m0 + r);
        xr[r][j] = (j < HID) ? hf[m * HID + j] : hb[m * HID + j - HID];
    }
    __syncthreads();
    int t = threadIdx.x, r = threadIdx.y;
    if (t < NT) {
        const float* w = Wp + t * 2 * HID;
        float acc = bp[t];
        for (int j = 0; j < 2 * HID; ++j) acc += xr[r][j] * w[j];
        int m = m0 + r;
        int s = m / BS, b = m % BS;
        E[((size_t)b * SL + s) * NT + t] = acc;
    }
}

// ---------------------------------------------------------------------------
// 6) CRF log-likelihood: one wave per batch element; lane = tag.
// ---------------------------------------------------------------------------
__global__ void crf_fwd_k(const float* __restrict__ E, const int* __restrict__ labels,
                          const int* __restrict__ mask, const float* __restrict__ start,
                          const float* __restrict__ endv, const float* __restrict__ trans,
                          float* __restrict__ lossp)
{
    __shared__ float tr[NT * NT];
    int b = blockIdx.x, j = threadIdx.x;
    for (int i = j; i < NT * NT; i += 32) tr[i] = trans[i];
    __syncthreads();

    const float NEG = -1e30f;
    float alpha = (j < NT) ? start[j] + E[((size_t)b * SL) * NT + j] : NEG;

    for (int s = 1; s < SL; ++s) {
        float mx = NEG;
        for (int i = 0; i < NT; ++i) {
            float ai = __shfl(alpha, i, 32);
            float v  = ai + ((j < NT) ? tr[i * NT + j] : 0.f);
            mx = fmaxf(mx, v);
        }
        float se = 0.f;
        for (int i = 0; i < NT; ++i) {
            float ai = __shfl(alpha, i, 32);
            float v  = ai + ((j < NT) ? tr[i * NT + j] : 0.f);
            se += __expf(v - mx);
        }
        float na = mx + __logf(se) + ((j < NT) ? E[((size_t)b * SL + s) * NT + j] : 0.f);
        if (mask[s * BS + b] && j < NT) alpha = na;
    }
    float zv = (j < NT) ? alpha + endv[j] : NEG;
    float mz = zv;
    for (int off = 16; off; off >>= 1) mz = fmaxf(mz, __shfl_xor(mz, off, 32));
    float sz = (j < NT) ? __expf(zv - mz) : 0.f;
    for (int off = 16; off; off >>= 1) sz += __shfl_xor(sz, off, 32);
    float Z = mz + __logf(sz);

    if (j == 0) {
        int y0 = labels[b];
        float num = start[y0] + E[((size_t)b * SL) * NT + y0];
        int yprev = y0, last = y0;
        for (int s = 1; s < SL; ++s) {
            int ys = labels[s * BS + b];
            if (mask[s * BS + b]) {
                num += tr[yprev * NT + ys] + E[((size_t)b * SL + s) * NT + ys];
                last = ys;
            }
            yprev = ys;
        }
        num += endv[last];
        lossp[b] = num - Z;
    }
}

__global__ void loss_reduce_k(const float* __restrict__ lossp, float* __restrict__ out) {
    __shared__ float s[64];
    int t = threadIdx.x;
    s[t] = lossp[t];
    __syncthreads();
    for (int off = 32; off; off >>= 1) { if (t < off) s[t] += s[t + off]; __syncthreads(); }
    if (t == 0) out[0] = -s[0];
}

// ---------------------------------------------------------------------------
// 7) Viterbi decode: one wave per batch element; backpointers to workspace.
// ---------------------------------------------------------------------------
__global__ void viterbi_k(const float* __restrict__ E, const int* __restrict__ mask,
                          const float* __restrict__ start, const float* __restrict__ endv,
                          const float* __restrict__ trans, int* __restrict__ bptr,
                          float* __restrict__ outPred)
{
    __shared__ float tr[NT * NT];
    __shared__ float fin[32];
    int b = blockIdx.x, j = threadIdx.x;
    for (int i = j; i < NT * NT; i += 32) tr[i] = trans[i];
    __syncthreads();

    const float NEG = -1e30f;
    float score = (j < NT) ? start[j] + E[((size_t)b * SL) * NT + j] : NEG;

    for (int s = 1; s < SL; ++s) {
        float best = NEG; int bi = 0;
        for (int i = 0; i < NT; ++i) {
            float si = __shfl(score, i, 32);
            float v  = si + ((j < NT) ? tr[i * NT + j] : 0.f);
            if (v > best) { best = v; bi = i; }
        }
        if (j < NT) bptr[((size_t)s * BS + b) * NT + j] = bi;
        if (mask[s * BS + b] && j < NT)
            score = best + E[((size_t)b * SL + s) * NT + j];
    }
    fin[j] = (j < NT) ? score + endv[j] : NEG;
    __syncthreads();

    if (j == 0) {
        int tag = 0; float bv = fin[0];
        for (int i = 1; i < NT; ++i) if (fin[i] > bv) { bv = fin[i]; tag = i; }
        for (int s = SL - 1; s >= 1; --s) {
            outPred[1 + (size_t)b * SL + s] = mask[s * BS + b] ? (float)tag : 0.f;
            if (mask[s * BS + b]) tag = bptr[((size_t)s * BS + b) * NT + tag];
        }
        outPred[1 + (size_t)b * SL] = mask[b] ? (float)tag : 0.f;
    }
}

// ---------------------------------------------------------------------------
// Launch
// ---------------------------------------------------------------------------
extern "C" void kernel_launch(void* const* d_in, const int* in_sizes, int n_in,
                              void* d_out, int out_size, void* d_ws, size_t ws_size,
                              hipStream_t stream)
{
    const float* word  = (const float*)d_in[0];
    const int*   mask  = (const int*)d_in[1];
    const int*   labels= (const int*)d_in[2];
    const float* wihf  = (const float*)d_in[3];
    const float* whhf  = (const float*)d_in[4];
    const float* bihf  = (const float*)d_in[5];
    const float* bhhf  = (const float*)d_in[6];
    const float* wihb  = (const float*)d_in[7];
    const float* whhb  = (const float*)d_in[8];
    const float* bihb  = (const float*)d_in[9];
    const float* bhhb  = (const float*)d_in[10];
    const float* gamma = (const float*)d_in[11];
    const float* beta  = (const float*)d_in[12];
    const float* linw  = (const float*)d_in[13];
    const float* linb  = (const float*)d_in[14];
    const float* cstart= (const float*)d_in[15];
    const float* cend  = (const float*)d_in[16];
    const float* ctrans= (const float*)d_in[17];
    float* out = (float*)d_out;

    char* ws = (char*)d_ws;
    size_t oXBF  = 0;
    size_t oWIHF = oXBF  + (size_t)SLBS * DIM * 2;
    size_t oWIHB = oWIHF + (size_t)GG * DIM * 2;
    size_t oWHHF = oWIHB + (size_t)GG * DIM * 2;
    size_t oWHHB = oWHHF + (size_t)GG * HID * 2;
    size_t oXGF  = oWHHB + (size_t)GG * HID * 2;
    size_t oXGB  = oXGF  + (size_t)SLBS * GG * 4;
    size_t oHF   = oXGB  + (size_t)SLBS * GG * 4;
    size_t oHB   = oHF   + (size_t)SLBS * HID * 4;
    size_t oMEAN = oHB   + (size_t)SLBS * HID * 4;
    size_t oVAR  = oMEAN + 2 * HID * 4;
    size_t oWP   = oVAR  + 2 * HID * 4;
    size_t oBP   = oWP   + (size_t)NT * 2 * HID * 4;
    size_t oE    = oBP   + 256;
    size_t oBPTR = oE    + (size_t)BS * SL * NT * 4;
    size_t oLP   = oBPTR + (size_t)SL * BS * NT * 4;

    unsigned short* Xbf   = (unsigned short*)(ws + oXBF);
    unsigned short* WihfB = (unsigned short*)(ws + oWIHF);
    unsigned short* WihbB = (unsigned short*)(ws + oWIHB);
    unsigned short* WhhfB = (unsigned short*)(ws + oWHHF);
    unsigned short* WhhbB = (unsigned short*)(ws + oWHHB);
    float* XGF  = (float*)(ws + oXGF);
    float* XGB  = (float*)(ws + oXGB);
    float* HF   = (float*)(ws + oHF);
    float* HB   = (float*)(ws + oHB);
    float* MEAN = (float*)(ws + oMEAN);
    float* VAR  = (float*)(ws + oVAR);
    float* WP   = (float*)(ws + oWP);
    float* BP   = (float*)(ws + oBP);
    float* Emat = (float*)(ws + oE);
    int*   BPTR = (int*)(ws + oBPTR);
    float* LP   = (float*)(ws + oLP);

    f32_to_bf16_k<<<4096, 256, 0, stream>>>(word, Xbf,   SLBS * DIM);
    f32_to_bf16_k<<<2048, 256, 0, stream>>>(wihf, WihfB, GG * DIM);
    f32_to_bf16_k<<<2048, 256, 0, stream>>>(wihb, WihbB, GG * DIM);
    f32_to_bf16_k<<<1024, 256, 0, stream>>>(whhf, WhhfB, GG * HID);
    f32_to_bf16_k<<<1024, 256, 0, stream>>>(whhb, WhhbB, GG * HID);

    dim3 ggrid(GG / 128, SLBS / 128);
    xg_gemm_k<<<ggrid, 256, 0, stream>>>(Xbf, WihfB, bihf, bhhf, XGF);
    xg_gemm_k<<<ggrid, 256, 0, stream>>>(Xbf, WihbB, bihb, bhhb, XGB);

    lstm_scan_k<<<2, 1024, 0, stream>>>(XGF, XGB, WhhfB, WhhbB, HF, HB);

    bn_stats_k<<<2 * HID, 256, 0, stream>>>(HF, HB, MEAN, VAR);
    fold_bn_k<<<NT, 256, 0, stream>>>(linw, linb, gamma, beta, MEAN, VAR, WP, BP);
    logits_k<<<SLBS / 8, dim3(32, 8), 0, stream>>>(HF, HB, WP, BP, Emat);

    crf_fwd_k<<<BS, 32, 0, stream>>>(Emat, labels, mask, cstart, cend, ctrans, LP);
    loss_reduce_k<<<1, 64, 0, stream>>>(LP, out);
    viterbi_k<<<BS, 32, 0, stream>>>(Emat, mask, cstart, cend, ctrans, BPTR, out);
}
